// GraphNet_51857435132406
// MI455X (gfx1250) — compile-verified, matrix operand order checked by reading, not run on hardware
//
#include <hip/hip_runtime.h>
#include <hip/hip_bf16.h>

// ---- problem constants (match reference) ----
#define N_NODES 16384
#define EDGES   400000
#define F_INP   512
#define HID     256
#define RELS    16
#define NBASES  30
#define NCLS    7
#define DDIM    768     // F_INP + HID
#define LSEQ    512
#define BDIAL   32

typedef __bf16 bf16;
typedef __attribute__((ext_vector_type(16))) __bf16 v16bf;
typedef __attribute__((ext_vector_type(8)))  __bf16 v8bf;
typedef __attribute__((ext_vector_type(8)))  float  v8f;

// 16-bit A/B-fragment loader (ISA 7.12.2, wave32): tile is 16x32 row-major
// at p (row stride ld). lane row = lane&15; lanes 0-15 hold K={0..7,16..23},
// lanes 16-31 hold K={8..15,24..31}.  Caller pre-applies row*ld + kb.
__device__ __forceinline__ v16bf load_frag(const bf16* __restrict__ p)
{
    union { v16bf v; v8bf h[2]; } u;
    u.h[0] = *(const v8bf*)(p);
    u.h[1] = *(const v8bf*)(p + 16);
    return u.v;
}

// =====================================================================
// Batched WMMA GEMM:  C[z] = A[z] @ Bt[z]^T   (bf16 in, f32 out)
//   A  : M x K row-major, leading dim lda, batch stride sA (elements)
//   Bt : N x K row-major (B pre-transposed), ldb, stride sB
//   C  : M x N row-major f32, ldc, stride sC
// One wave computes a 64x32 output tile (4 M-subtiles x 2 N-subtiles):
// 8 v_wmma per K-step from 12 b128 loads.
// grid = (N/32, M/64, batch), block = 32.
// Requires M%64==0, N%32==0, K%32==0 (true for all calls here).
// =====================================================================
__global__ __launch_bounds__(32)
void k_gemm(const bf16* __restrict__ A, const bf16* __restrict__ Bt,
            float* __restrict__ C, int K, int lda, int ldb, int ldc,
            long long sA, long long sB, long long sC)
{
    const int lane = threadIdx.x;
    const int m0 = blockIdx.y << 6;
    const int n0 = blockIdx.x << 5;
    A  += (long long)blockIdx.z * sA + (long long)m0 * lda;
    Bt += (long long)blockIdx.z * sB + (long long)n0 * ldb;
    C  += (long long)blockIdx.z * sC;

    const int row = lane & 15;
    const int kb  = (lane >> 4) << 3;          // 0 or 8
    const bf16* pa = A  + (long long)row * lda + kb;
    const bf16* pb = Bt + (long long)row * ldb + kb;
    const long long stepA = (long long)16 * lda;   // next M-subtile
    const long long stepB = (long long)16 * ldb;   // next N-subtile

    v8f acc[4][2] = {};
#pragma unroll 2
    for (int k = 0; k < K; k += 32) {
        // prefetch 2 K-steps ahead (global_prefetch_b8)
        __builtin_prefetch((const void*)(pa + k + 64), 0, 0);
        __builtin_prefetch((const void*)(pb + k + 64), 0, 0);

        v16bf a[4], b[2];
#pragma unroll
        for (int i = 0; i < 4; ++i) a[i] = load_frag(pa + i * stepA + k);
#pragma unroll
        for (int j = 0; j < 2; ++j) b[j] = load_frag(pb + j * stepB + k);
#pragma unroll
        for (int i = 0; i < 4; ++i)
#pragma unroll
            for (int j = 0; j < 2; ++j)
                acc[i][j] = __builtin_amdgcn_wmma_f32_16x16x32_bf16(
                    false, a[i], false, b[j], (short)0, acc[i][j],
                    false, false);
    }

    // C layout per 16x16 tile: VGPR v -> M = v + 8*(lane>=16), N = lane&15
    const int mb  = (lane >> 4) << 3;
    const int col = lane & 15;
#pragma unroll
    for (int i = 0; i < 4; ++i) {
#pragma unroll
        for (int j = 0; j < 2; ++j) {
            float* cp = C + (long long)(m0 + i * 16 + mb) * ldc
                          + (n0 + j * 16 + col);
#pragma unroll
            for (int v = 0; v < 8; ++v)
                cp[(long long)v * ldc] = acc[i][j][v];
        }
    }
}

// ---- fp32 -> bf16 elementwise convert ----
__global__ void k_cvt(const float* __restrict__ s, bf16* __restrict__ d,
                      long long n)
{
    long long i = (long long)blockIdx.x * blockDim.x + threadIdx.x;
    if (i < n) d[i] = (bf16)s[i];
}

// ---- transpose (rows x cols f32) -> (cols x rows bf16) ----
__global__ void k_tpose(const float* __restrict__ src, bf16* __restrict__ dst,
                        int rows, int cols)
{
    long long i = (long long)blockIdx.x * blockDim.x + threadIdx.x;
    if (i >= (long long)rows * cols) return;
    int cc = (int)(i % cols), rr = (int)(i / cols);
    dst[(long long)cc * rows + rr] = (bf16)src[i];
}

// ---- Wt[r][o][i] = sum_b att[r][b] * basis[b][i][o]  (bf16, pre-transposed)
__global__ void k_basis(const float* __restrict__ att,
                        const float* __restrict__ basis,
                        bf16* __restrict__ wt)
{
    long long idx = (long long)blockIdx.x * blockDim.x + threadIdx.x;
    if (idx >= (long long)RELS * HID * F_INP) return;
    int i = (int)(idx % F_INP);
    int o = (int)((idx / F_INP) % HID);
    int r = (int)(idx / ((long long)F_INP * HID));
    float s = 0.f;
#pragma unroll 5
    for (int b = 0; b < NBASES; ++b)
        s += att[r * NBASES + b] * basis[((long long)b * F_INP + i) * HID + o];
    wt[idx] = (bf16)s;                 // layout: r*(HID*F_INP) + o*F_INP + i
}

// ---- per-destination edge count ----
__global__ void k_count(const int* __restrict__ dstv, float* __restrict__ cnt)
{
    int e = blockIdx.x * blockDim.x + threadIdx.x;
    if (e < EDGES) atomicAdd(&cnt[dstv[e]], 1.0f);
}

// ---- RGCN message scatter for one relation (8 floats per thread) ----
__global__ void k_scat_rel(const float* __restrict__ xw,
                           const int* __restrict__ srcv,
                           const int* __restrict__ dstv,
                           const int* __restrict__ etype,
                           const float* __restrict__ enorm,
                           float* __restrict__ msum, int rel)
{
    long long tid = (long long)blockIdx.x * blockDim.x + threadIdx.x;
    int e = (int)(tid >> 5);
    if (e >= EDGES) return;
    if (etype[e] != rel) return;
    int h0 = ((int)tid & 31) << 3;
    int s = srcv[e], d = dstv[e];
    float w = enorm[e];
    const float* xp = xw + (long long)s * HID + h0;
    float* mp = msum + (long long)d * HID + h0;
#pragma unroll
    for (int i = 0; i < 8; ++i) atomicAdd(mp + i, xp[i] * w);
}

// ---- GraphConv scatter-add ----
__global__ void k_scat_add(const float* __restrict__ hbuf,
                           const int* __restrict__ srcv,
                           const int* __restrict__ dstv,
                           float* __restrict__ aggr2)
{
    long long tid = (long long)blockIdx.x * blockDim.x + threadIdx.x;
    int e = (int)(tid >> 5);
    if (e >= EDGES) return;
    int h0 = ((int)tid & 31) << 3;
    int s = srcv[e], d = dstv[e];
    const float* xp = hbuf + (long long)s * HID + h0;
    float* mp = aggr2 + (long long)d * HID + h0;
#pragma unroll
    for (int i = 0; i < 8; ++i) atomicAdd(mp + i, xp[i]);
}

// ---- out1 = mean-aggr + x@root + bias  (emit bf16 for next GEMMs) ----
__global__ void k_out1(const float* __restrict__ msum,
                       const float* __restrict__ cnt,
                       const float* __restrict__ xr,
                       const float* __restrict__ rb,
                       bf16* __restrict__ out1bf)
{
    long long idx = (long long)blockIdx.x * blockDim.x + threadIdx.x;
    if (idx >= (long long)N_NODES * HID) return;
    int n = (int)(idx >> 8);
    int h = (int)idx & (HID - 1);
    float c = cnt[n];
    float a = (c > 0.f) ? msum[idx] / fmaxf(c, 1.f) : 0.f;
    out1bf[idx] = (bf16)(a + xr[idx] + rb[h]);
}

// ---- emotions = concat(x, out2); emit plain / masked / per-batch-transposed
__global__ void k_emotions(const float* __restrict__ x,
                           const float* __restrict__ aggr2,
                           const float* __restrict__ glbuf,
                           const float* __restrict__ gclb,
                           const float* __restrict__ umask,
                           bf16* __restrict__ emb,
                           bf16* __restrict__ embm,
                           bf16* __restrict__ embT)
{
    long long idx = (long long)blockIdx.x * blockDim.x + threadIdx.x;
    if (idx >= (long long)N_NODES * DDIM) return;
    int n = (int)(idx / DDIM);
    int d = (int)(idx % DDIM);
    float v;
    if (d < F_INP) {
        v = x[(long long)n * F_INP + d];
    } else {
        long long j = (long long)n * HID + (d - F_INP);
        v = aggr2[j] + glbuf[j] + gclb[d - F_INP];
    }
    int b = n >> 9;            // n / LSEQ
    int t = n & (LSEQ - 1);
    float um = umask[n];       // umask[b*LSEQ + t] == umask[n]
    emb[idx]  = (bf16)v;
    embm[idx] = (bf16)(v * um);
    embT[((long long)b * DDIM + d) * LSEQ + t] = (bf16)v;
}

// ---- Xc_bf = bf16(Xc_f + mt_b) ----
__global__ void k_xcbias(const float* __restrict__ xcf,
                         const float* __restrict__ mtb,
                         bf16* __restrict__ xcbf)
{
    long long idx = (long long)blockIdx.x * blockDim.x + threadIdx.x;
    if (idx >= (long long)N_NODES * DDIM) return;
    xcbf[idx] = (bf16)(xcf[idx] + mtb[idx % DDIM]);
}

// ---- masked tanh-softmax over s (one block per (b,t) row) ----
__global__ __launch_bounds__(256)
void k_softmax(const float* __restrict__ scores,
               const float* __restrict__ umask,
               bf16* __restrict__ alpha)
{
    __shared__ float sv[LSEQ];
    __shared__ float red[256];
    const int row = blockIdx.x;                // b*LSEQ + t
    const int b   = row >> 9;
    const float* sp = scores + (long long)row * LSEQ;
    const float* um = umask + (long long)b * LSEQ;
    const int t0 = threadIdx.x;

    float lmax = -1e30f;
    for (int s = t0; s < LSEQ; s += 256) {
        float v = tanhf(sp[s] * um[s]);
        sv[s] = v;
        lmax = fmaxf(lmax, v);
    }
    red[t0] = lmax; __syncthreads();
    for (int o = 128; o > 0; o >>= 1) {
        if (t0 < o) red[t0] = fmaxf(red[t0], red[t0 + o]);
        __syncthreads();
    }
    const float mx = red[0]; __syncthreads();

    float lsum = 0.f;
    for (int s = t0; s < LSEQ; s += 256) {
        float e = __expf(sv[s] - mx) * um[s];   // softmax * mask (Z cancels)
        sv[s] = e; lsum += e;
    }
    red[t0] = lsum; __syncthreads();
    for (int o = 128; o > 0; o >>= 1) {
        if (t0 < o) red[t0] += red[t0 + o];
        __syncthreads();
    }
    const float inv = 1.0f / red[0];
    bf16* ap = alpha + (long long)row * LSEQ;
    for (int s = t0; s < LSEQ; s += 256) ap[s] = (bf16)(sv[s] * inv);
}

// ---- classifier tail: relu(hid+b) @ fc_w + fc_b -> log_softmax ----
__global__ void k_logits(const float* __restrict__ hid,
                         const float* __restrict__ lin_b,
                         const float* __restrict__ fc_w,
                         const float* __restrict__ fc_b,
                         float* __restrict__ out)
{
    int n = blockIdx.x * blockDim.x + threadIdx.x;
    if (n >= N_NODES) return;
    float acc[NCLS];
#pragma unroll
    for (int c = 0; c < NCLS; ++c) acc[c] = fc_b[c];
    const float* hp = hid + (long long)n * HID;
    for (int j = 0; j < HID; ++j) {
        float h = fmaxf(hp[j] + lin_b[j], 0.f);
#pragma unroll
        for (int c = 0; c < NCLS; ++c) acc[c] += h * fc_w[j * NCLS + c];
    }
    float mx = acc[0];
#pragma unroll
    for (int c = 1; c < NCLS; ++c) mx = fmaxf(mx, acc[c]);
    float s = 0.f;
#pragma unroll
    for (int c = 0; c < NCLS; ++c) s += __expf(acc[c] - mx);
    float lse = mx + __logf(s);
#pragma unroll
    for (int c = 0; c < NCLS; ++c) out[(long long)n * NCLS + c] = acc[c] - lse;
}

// =====================================================================
extern "C" void kernel_launch(void* const* d_in, const int* in_sizes, int n_in,
                              void* d_out, int out_size, void* d_ws, size_t ws_size,
                              hipStream_t stream)
{
    (void)in_sizes; (void)n_in; (void)out_size; (void)ws_size;

    const float* x      = (const float*)d_in[0];
    const int*   eidx   = (const int*)  d_in[1];
    const int*   esrc   = eidx;
    const int*   edst   = eidx + EDGES;
    const float* enorm  = (const float*)d_in[2];
    const int*   etype  = (const int*)  d_in[3];
    const float* umask  = (const float*)d_in[5];
    const float* basis  = (const float*)d_in[8];
    const float* attw   = (const float*)d_in[9];
    const float* root   = (const float*)d_in[10];
    const float* rgcn_b = (const float*)d_in[11];
    const float* gc_w   = (const float*)d_in[12];
    const float* gc_lw  = (const float*)d_in[13];
    const float* gc_lb  = (const float*)d_in[14];
    const float* mt_w   = (const float*)d_in[15];
    const float* mt_b   = (const float*)d_in[16];
    const float* lin_w  = (const float*)d_in[17];
    const float* lin_b  = (const float*)d_in[18];
    const float* fc_w   = (const float*)d_in[19];
    const float* fc_b   = (const float*)d_in[20];
    float* out = (float*)d_out;

    // ---- workspace carve-out (256B aligned) ----
    char* w = (char*)d_ws;
    size_t off = 0;
    auto alloc = [&](size_t bytes) -> void* {
        void* p = w + off;
        off = (off + bytes + 255) & ~(size_t)255;
        return p;
    };
    bf16*  Xbf    = (bf16*)alloc((size_t)N_NODES * F_INP * 2);
    bf16*  WtBf   = (bf16*)alloc((size_t)RELS * HID * F_INP * 2);
    bf16*  rootT  = (bf16*)alloc((size_t)HID * F_INP * 2);
    bf16*  gcwT   = (bf16*)alloc((size_t)HID * HID * 2);
    bf16*  gclinT = (bf16*)alloc((size_t)HID * HID * 2);
    bf16*  mtT    = (bf16*)alloc((size_t)DDIM * DDIM * 2);
    bf16*  linT   = (bf16*)alloc((size_t)HID * DDIM * 2);
    float* XWr    = (float*)alloc((size_t)N_NODES * HID * 4);   // reused as glbuf
    float* msum   = (float*)alloc((size_t)N_NODES * HID * 4);
    float* cnt    = (float*)alloc((size_t)N_NODES * 4);
    float* xr     = (float*)alloc((size_t)N_NODES * HID * 4);
    bf16*  out1bf = (bf16*)alloc((size_t)N_NODES * HID * 2);
    float* hbuf   = (float*)alloc((size_t)N_NODES * HID * 4);
    float* aggr2  = (float*)alloc((size_t)N_NODES * HID * 4);
    bf16*  emb    = (bf16*)alloc((size_t)N_NODES * DDIM * 2);
    bf16*  embm   = (bf16*)alloc((size_t)N_NODES * DDIM * 2);
    bf16*  embT   = (bf16*)alloc((size_t)BDIAL * DDIM * LSEQ * 2);
    float* XcF    = (float*)alloc((size_t)N_NODES * DDIM * 4);  // reused as att_em
    bf16*  XcBf   = (bf16*)alloc((size_t)N_NODES * DDIM * 2);
    float* scores = (float*)alloc((size_t)BDIAL * LSEQ * LSEQ * 4);
    bf16*  alpha  = (bf16*)alloc((size_t)BDIAL * LSEQ * LSEQ * 2);
    bf16*  aembBf = (bf16*)alloc((size_t)N_NODES * DDIM * 2);
    float* hidF   = (float*)alloc((size_t)N_NODES * HID * 4);

    auto gemm = [&](const bf16* A, const bf16* Bt, float* C, int M, int N,
                    int K, int lda, int ldb, int ldc, int batch,
                    long long sA, long long sB, long long sC) {
        dim3 g((unsigned)(N / 32), (unsigned)(M / 64), (unsigned)batch);
        k_gemm<<<g, 32, 0, stream>>>(A, Bt, C, K, lda, ldb, ldc, sA, sB, sC);
    };

    // ---- stage 0: precision conversion / weight prep ----
    k_cvt<<<(N_NODES * F_INP + 255) / 256, 256, 0, stream>>>(
        x, Xbf, (long long)N_NODES * F_INP);
    k_basis<<<(RELS * HID * F_INP + 255) / 256, 256, 0, stream>>>(
        attw, basis, WtBf);
    k_tpose<<<(F_INP * HID + 255) / 256, 256, 0, stream>>>(root, rootT, F_INP, HID);
    k_tpose<<<(HID * HID + 255) / 256, 256, 0, stream>>>(gc_w, gcwT, HID, HID);
    k_tpose<<<(HID * HID + 255) / 256, 256, 0, stream>>>(gc_lw, gclinT, HID, HID);
    k_tpose<<<(DDIM * DDIM + 255) / 256, 256, 0, stream>>>(mt_w, mtT, DDIM, DDIM);
    k_tpose<<<(DDIM * HID + 255) / 256, 256, 0, stream>>>(lin_w, linT, DDIM, HID);

    hipMemsetAsync(msum, 0, (size_t)N_NODES * HID * 4, stream);
    hipMemsetAsync(cnt, 0, (size_t)N_NODES * 4, stream);
    hipMemsetAsync(aggr2, 0, (size_t)N_NODES * HID * 4, stream);

    // ---- RGCNConv ----
    gemm(Xbf, rootT, xr, N_NODES, HID, F_INP, F_INP, F_INP, HID, 1, 0, 0, 0);
    k_count<<<(EDGES + 255) / 256, 256, 0, stream>>>(edst, cnt);

    const long long scat_threads = (long long)EDGES * 32;
    const int scat_blocks = (int)((scat_threads + 255) / 256);
    for (int r = 0; r < RELS; ++r) {
        gemm(Xbf, WtBf + (size_t)r * HID * F_INP, XWr,
             N_NODES, HID, F_INP, F_INP, F_INP, HID, 1, 0, 0, 0);
        k_scat_rel<<<scat_blocks, 256, 0, stream>>>(
            XWr, esrc, edst, etype, enorm, msum, r);
    }
    k_out1<<<(N_NODES * HID + 255) / 256, 256, 0, stream>>>(
        msum, cnt, xr, rgcn_b, out1bf);

    // ---- GraphConv ----
    gemm(out1bf, gcwT, hbuf, N_NODES, HID, HID, HID, HID, HID, 1, 0, 0, 0);
    k_scat_add<<<scat_blocks, 256, 0, stream>>>(hbuf, esrc, edst, aggr2);
    float* glbuf = XWr;  // relation loop is done; reuse
    gemm(out1bf, gclinT, glbuf, N_NODES, HID, HID, HID, HID, HID, 1, 0, 0, 0);

    // ---- emotions (plain / masked / per-batch transposed) ----
    k_emotions<<<(int)(((long long)N_NODES * DDIM + 255) / 256), 256, 0, stream>>>(
        x, aggr2, glbuf, gc_lb, umask, emb, embm, embT);

    // ---- MatchingAttention ----
    gemm(emb, mtT, XcF, N_NODES, DDIM, DDIM, DDIM, DDIM, DDIM, 1, 0, 0, 0);
    k_xcbias<<<(int)(((long long)N_NODES * DDIM + 255) / 256), 256, 0, stream>>>(
        XcF, mt_b, XcBf);
    // scores[b] = Xc_b (L x D) @ Mm_b^T  -> (L x L)
    gemm(XcBf, embm, scores, LSEQ, LSEQ, DDIM, DDIM, DDIM, LSEQ, BDIAL,
         (long long)LSEQ * DDIM, (long long)LSEQ * DDIM, (long long)LSEQ * LSEQ);
    k_softmax<<<BDIAL * LSEQ, 256, 0, stream>>>(scores, umask, alpha);
    // att_em[b] = alpha_b (L x L) @ M_b (L x D)  (Bt = per-batch transposed M)
    float* attem = XcF;  // reuse
    gemm(alpha, embT, attem, LSEQ, DDIM, LSEQ, LSEQ, LSEQ, DDIM, BDIAL,
         (long long)LSEQ * LSEQ, (long long)DDIM * LSEQ, (long long)LSEQ * DDIM);
    k_cvt<<<(int)(((long long)N_NODES * DDIM + 255) / 256), 256, 0, stream>>>(
        attem, aembBf, (long long)N_NODES * DDIM);

    // ---- classifier ----
    gemm(aembBf, linT, hidF, N_NODES, HID, DDIM, DDIM, DDIM, HID, 1, 0, 0, 0);
    k_logits<<<(N_NODES + 255) / 256, 256, 0, stream>>>(
        hidF, lin_b, fc_w, fc_b, out);
}
